// DetectionLoss_76321568850056
// MI455X (gfx1250) — compile-verified, matrix operand order checked by reading, not run on hardware
//
#include <hip/hip_runtime.h>
#include <hip/hip_bf16.h>

// Problem constants (match reference setup_inputs)
#define BB 2
#define NN 256
#define MM 32
#define LL 16
#define VV 32000
#define BIGC 1e9f
#define EPSC 1e-7f

typedef __attribute__((ext_vector_type(16))) _Float16 v16h;
typedef __attribute__((ext_vector_type(8)))  float    v8f;

// ---------------------------------------------------------------------------
// GIoU for a prediction/gt box pair (with x1<=x2, y1<=y2 normalization)
// ---------------------------------------------------------------------------
__device__ __forceinline__ float giou_pair(float4 p, float4 g) {
    float ax1 = fminf(p.x, p.z), ay1 = fminf(p.y, p.w);
    float ax2 = fmaxf(p.x, p.z), ay2 = fmaxf(p.y, p.w);
    float bx1 = fminf(g.x, g.z), by1 = fminf(g.y, g.w);
    float bx2 = fmaxf(g.x, g.z), by2 = fmaxf(g.y, g.w);
    float xi1 = fmaxf(ax1, bx1), yi1 = fmaxf(ay1, by1);
    float xi2 = fminf(ax2, bx2), yi2 = fminf(ay2, by2);
    float inter = fmaxf(xi2 - xi1, 0.f) * fmaxf(yi2 - yi1, 0.f);
    float a1 = (ax2 - ax1) * (ay2 - ay1);
    float a2 = (bx2 - bx1) * (by2 - by1);
    float uni = a1 + a2 - inter;
    float iou = inter / (uni + EPSC);
    float xe1 = fminf(ax1, bx1), ye1 = fminf(ay1, by1);
    float xe2 = fmaxf(ax2, bx2), ye2 = fmaxf(ay2, by2);
    float enc = (xe2 - xe1) * (ye2 - ye1);
    return iou - (enc - uni) / (enc + EPSC);
}

// ---------------------------------------------------------------------------
// Kernel A: pairwise cost matrix (B, N, M)
// ---------------------------------------------------------------------------
__global__ __launch_bounds__(NN) void cost_kernel(
        const float* __restrict__ pb, const float* __restrict__ po,
        const float* __restrict__ gb, float* __restrict__ cost) {
    int b = blockIdx.x;
    int n = threadIdx.x;
    float4 p = *(const float4*)(pb + ((size_t)(b * NN + n)) * 4);
    float x = po[b * NN + n];
    float objc = 1.f - 1.f / (1.f + __expf(-x));   // 1 - sigmoid
    float* out = cost + ((size_t)(b * NN + n)) * MM;
    for (int m = 0; m < MM; ++m) {
        float4 g = *(const float4*)(gb + ((size_t)(b * MM + m)) * 4);
        float l1 = fabsf(p.x - g.x) + fabsf(p.y - g.y) +
                   fabsf(p.z - g.z) + fabsf(p.w - g.w);
        float gi = giou_pair(p, g);
        out[m] = l1 + (1.f - gi) + objc;
    }
}

// ---------------------------------------------------------------------------
// Kernel B: greedy matching (replicates jax argmin scan), bbox + obj losses
// ---------------------------------------------------------------------------
__device__ __forceinline__ float block_sum256(float* s_val, int t) {
    for (int s = 128; s > 0; s >>= 1) {
        if (t < s) s_val[t] += s_val[t + s];
        __syncthreads();
    }
    return s_val[0];
}

__global__ __launch_bounds__(NN) void match_kernel(
        const float* __restrict__ pb, const float* __restrict__ po,
        const float* __restrict__ gb, const float* __restrict__ cost,
        int* __restrict__ pis, int* __restrict__ gjs,
        float* __restrict__ bbox_out, float* __restrict__ obj_out) {
    __shared__ float s_val[NN];
    __shared__ int   s_idx[NN];
    __shared__ int   usedRow[NN];
    __shared__ int   usedCol[MM];
    __shared__ int   s_pi[MM], s_gj[MM];
    __shared__ float s_l1;

    int b = blockIdx.x;
    int t = threadIdx.x;
    const float* C = cost + (size_t)b * NN * MM;

    usedRow[t] = 0;
    if (t < MM) usedCol[t] = 0;
    __syncthreads();

    // 32 sequential global-argmin picks; +1e9 masking like the reference,
    // tie-break on smallest row-major flat index (jnp.argmin semantics).
    for (int step = 0; step < MM; ++step) {
        float rowpen = usedRow[t] ? BIGC : 0.f;
        float best = 3.0e38f;
        int bidx = 0x7fffffff;
        for (int m = 0; m < MM; ++m) {
            float v = C[t * MM + m] + rowpen + (usedCol[m] ? BIGC : 0.f);
            if (v < best) { best = v; bidx = t * MM + m; }
        }
        s_val[t] = best; s_idx[t] = bidx;
        __syncthreads();
        for (int s = 128; s > 0; s >>= 1) {
            if (t < s) {
                float v2 = s_val[t + s]; int i2 = s_idx[t + s];
                if (v2 < s_val[t] || (v2 == s_val[t] && i2 < s_idx[t])) {
                    s_val[t] = v2; s_idx[t] = i2;
                }
            }
            __syncthreads();
        }
        if (t == 0) {
            int f = s_idx[0];
            int n = f / MM, m = f - n * MM;
            s_pi[step] = n; s_gj[step] = m;
            usedRow[n] = 1; usedCol[m] = 1;
        }
        __syncthreads();
    }

    // matched-pair bbox losses
    float l1v = 0.f, gv = 0.f;
    if (t < MM) {
        float4 p = *(const float4*)(pb + ((size_t)(b * NN + s_pi[t])) * 4);
        float4 g = *(const float4*)(gb + ((size_t)(b * MM + s_gj[t])) * 4);
        l1v = fabsf(p.x - g.x) + fabsf(p.y - g.y) +
              fabsf(p.z - g.z) + fabsf(p.w - g.w);
        gv = 1.f - giou_pair(p, g);
        pis[b * MM + t] = s_pi[t];
        gjs[b * MM + t] = s_gj[t];
    }
    s_val[t] = (t < MM) ? l1v : 0.f;
    __syncthreads();
    float l1sum = block_sum256(s_val, t);
    if (t == 0) s_l1 = l1sum / (float)(MM * 4);
    __syncthreads();

    s_val[t] = (t < MM) ? gv : 0.f;
    __syncthreads();
    float gsum = block_sum256(s_val, t);
    if (t == 0) {
        float gl = fminf(fmaxf(gsum / (float)MM, 0.f), 2.f);
        bbox_out[b] = fmaxf(s_l1 + gl, 0.f);
    }
    __syncthreads();

    // objectness BCE-with-logits over all N predictions
    float x = po[b * NN + t];
    float tt = usedRow[t] ? 1.f : 0.f;
    s_val[t] = fmaxf(x, 0.f) - x * tt + log1pf(__expf(-fabsf(x)));
    __syncthreads();
    float osum = block_sum256(s_val, t);
    if (t == 0) obj_out[b] = fmaxf(osum / (float)NN, 0.f);
}

// ---------------------------------------------------------------------------
// Kernel C: caption CE — 960 softmax rows of V=32000 fp32.
// 60 blocks x 8 waves; each block owns 16 rows. Pass 1: segmented row max.
// Pass 2: exp-rowsum via v_wmma_f32_16x16x32_f16 against an all-ones B
// (flash-softmax rowsum trick; exp values in [0,1], f32 accumulation).
// ---------------------------------------------------------------------------
#define WPB 8   // waves per block

__global__ __launch_bounds__(32 * WPB) void cap_ce_kernel(
        const float* __restrict__ logits, const int* __restrict__ tok,
        const int* __restrict__ pis, const int* __restrict__ gjs,
        float* __restrict__ ce) {
    __shared__ float smax[WPB * 32];
    __shared__ float ssum[WPB * 32 * 8];

    int tid  = threadIdx.x;
    int wave = tid >> 5;
    int lane = tid & 31;
    int r    = lane & 15;      // row within this block's 16-row tile
    int hi   = lane >> 4;      // lane-half selector (WMMA A layout)

    int g   = blockIdx.x * 16 + r;          // global row id 0..959
    int b   = g / (MM * (LL - 1));
    int rem = g - b * (MM * (LL - 1));
    int m   = rem / (LL - 1);
    int l   = rem - m * (LL - 1);
    int pi  = pis[b * MM + m];
    int gj  = gjs[b * MM + m];
    const float* row = logits + ((size_t)((b * NN + pi) * LL + l)) * (size_t)VV;

    // ---- pass 1: row max; 16 threads per row, 2000 contiguous floats each
    const int SEG = VV / (2 * WPB);                 // 2000
    const float* seg = row + (hi * WPB + wave) * SEG;
    float mx = -3.0e38f;
    for (int i = 0; i < SEG; i += 4) {
        float4 v = *(const float4*)(seg + i);
        __builtin_prefetch(seg + i + 128, 0, 1);    // global_prefetch_b8
        mx = fmaxf(mx, fmaxf(fmaxf(v.x, v.y), fmaxf(v.z, v.w)));
    }
    smax[wave * 32 + lane] = mx;
    __syncthreads();
    float mfull = -3.0e38f;
    #pragma unroll
    for (int w = 0; w < WPB; ++w) {
        mfull = fmaxf(mfull, smax[w * 32 + r]);
        mfull = fmaxf(mfull, smax[w * 32 + r + 16]);
    }

    // ---- pass 2: sum(exp(x - max)) via WMMA rowsum (K-split across waves)
    v16h ones;
    #pragma unroll
    for (int j = 0; j < 16; ++j) ones[j] = (_Float16)1.0f;
    v8f acc = {0.f, 0.f, 0.f, 0.f, 0.f, 0.f, 0.f, 0.f};

    for (int k0 = wave * 32; k0 < VV; k0 += 32 * WPB) {
        // 16-bit A-matrix layout: lanes 0-15 hold K=[k0..k0+7],[k0+16..k0+23];
        // lanes 16-31 hold K=[k0+8..k0+15],[k0+24..k0+31].
        int base = k0 + hi * 8;
        float4 x0 = *(const float4*)(row + base);
        float4 x1 = *(const float4*)(row + base + 4);
        float4 x2 = *(const float4*)(row + base + 16);
        float4 x3 = *(const float4*)(row + base + 20);
        v16h a;
        a[0]  = (_Float16)__expf(x0.x - mfull); a[1]  = (_Float16)__expf(x0.y - mfull);
        a[2]  = (_Float16)__expf(x0.z - mfull); a[3]  = (_Float16)__expf(x0.w - mfull);
        a[4]  = (_Float16)__expf(x1.x - mfull); a[5]  = (_Float16)__expf(x1.y - mfull);
        a[6]  = (_Float16)__expf(x1.z - mfull); a[7]  = (_Float16)__expf(x1.w - mfull);
        a[8]  = (_Float16)__expf(x2.x - mfull); a[9]  = (_Float16)__expf(x2.y - mfull);
        a[10] = (_Float16)__expf(x2.z - mfull); a[11] = (_Float16)__expf(x2.w - mfull);
        a[12] = (_Float16)__expf(x3.x - mfull); a[13] = (_Float16)__expf(x3.y - mfull);
        a[14] = (_Float16)__expf(x3.z - mfull); a[15] = (_Float16)__expf(x3.w - mfull);
        // D = A(16x32 f16) x ones(32x16 f16) + C  -> every column = row sums
        acc = __builtin_amdgcn_wmma_f32_16x16x32_f16(
                  false, a, false, ones, (short)0, acc, false, false);
    }

    #pragma unroll
    for (int j = 0; j < 8; ++j) ssum[(wave * 32 + lane) * 8 + j] = acc[j];
    __syncthreads();

    if (wave == 0 && lane < 16) {
        // C/D layout: row r<8 -> VGPR r @ lanes 0-15; r>=8 -> VGPR r-8 @ lanes 16-31
        int sel = ((r < 8) ? 0 : 16) * 8 + (r & 7);
        float s = 0.f;
        #pragma unroll
        for (int w = 0; w < WPB; ++w) s += ssum[w * 256 + sel];
        int tgt = tok[(b * MM + gj) * LL + (l + 1)];
        ce[g] = (mfull + logf(s)) - row[tgt];       // logsumexp - x[target]
    }
}

// ---------------------------------------------------------------------------
// Kernel D: finalize -> out[4] = {total, 5*bbox.mean, 0.1*cap.mean, obj.mean}
// ---------------------------------------------------------------------------
__global__ __launch_bounds__(64) void finalize_kernel(
        const float* __restrict__ ce, const float* __restrict__ bbox,
        const float* __restrict__ obj, float* __restrict__ out) {
    __shared__ float caps[BB * MM];
    int t = threadIdx.x;            // 64 threads: (b, m) pairs
    if (t < BB * MM) {
        float s = 0.f;
        for (int l = 0; l < LL - 1; ++l) s += ce[t * (LL - 1) + l];
        caps[t] = fmaxf(s / (float)(LL - 1), 0.f);
    }
    __syncthreads();
    if (t == 0) {
        float cap0 = 0.f, cap1 = 0.f;
        for (int m = 0; m < MM; ++m) { cap0 += caps[m]; cap1 += caps[MM + m]; }
        cap0 = fmaxf(cap0 / (float)MM, 0.f);
        cap1 = fmaxf(cap1 / (float)MM, 0.f);
        float tot = 0.5f * ((5.0f * bbox[0] + 0.1f * cap0 + obj[0]) +
                            (5.0f * bbox[1] + 0.1f * cap1 + obj[1]));
        out[0] = fmaxf(tot, 0.f);
        out[1] = 5.0f  * 0.5f * (bbox[0] + bbox[1]);
        out[2] = 0.1f  * 0.5f * (cap0 + cap1);
        out[3] = 1.0f  * 0.5f * (obj[0] + obj[1]);
    }
}

// ---------------------------------------------------------------------------
// Launch
// ---------------------------------------------------------------------------
extern "C" void kernel_launch(void* const* d_in, const int* in_sizes, int n_in,
                              void* d_out, int out_size, void* d_ws, size_t ws_size,
                              hipStream_t stream) {
    const float* pb = (const float*)d_in[0];   // (B,N,4)
    const float* po = (const float*)d_in[1];   // (B,N)
    const float* cl = (const float*)d_in[2];   // (B,N,L,V)
    const float* gb = (const float*)d_in[3];   // (B,M,4)
    const int*   gt = (const int*)  d_in[4];   // (B,M,L)

    char* ws = (char*)d_ws;
    float* cost    = (float*)(ws + 0);          // B*N*M floats = 65536 B
    int*   pis     = (int*)  (ws + 65536);      // B*M ints
    int*   gjs     = (int*)  (ws + 65792);      // B*M ints
    float* bbox    = (float*)(ws + 66048);      // B floats
    float* obj     = (float*)(ws + 66056);      // B floats
    float* ce      = (float*)(ws + 66064);      // B*M*(L-1) floats

    cost_kernel<<<dim3(BB), dim3(NN), 0, stream>>>(pb, po, gb, cost);
    match_kernel<<<dim3(BB), dim3(NN), 0, stream>>>(pb, po, gb, cost,
                                                    pis, gjs, bbox, obj);
    cap_ce_kernel<<<dim3((BB * MM * (LL - 1)) / 16), dim3(32 * WPB), 0, stream>>>(
        cl, gt, pis, gjs, ce);
    finalize_kernel<<<dim3(1), dim3(64), 0, stream>>>(ce, bbox, obj, (float*)d_out);
}
